// MaskedViT_661424964084
// MI455X (gfx1250) — matrix operand order repaired and checked by reading, not compiled, not run on hardware
//
#include <hip/hip_runtime.h>
#include <math.h>

// ---------------- constants ----------------
constexpr int BB    = 32;     // batch
constexpr int IMG   = 224;
constexpr int PP    = 16;
constexpr int CC    = 3;
constexpr int GG    = 14;     // IMG/PP
constexpr int NTOK  = 197;    // GG*GG + 1
constexpr int NPAT  = 196;
constexpr int DM    = 768;
constexpr int NH    = 12;
constexpr int HD    = 64;
constexpr int HID   = 3072;
constexpr int GHD   = 128;
constexpr int NWARM = 2;
constexpr int NDYN  = 10;
constexpr int TOPK  = 118;
constexpr int NCLS  = 1000;
constexpr int TOKS  = BB * NTOK;        // 6304
constexpr float ATT_SCALE = 0.125f;     // 64^-0.5
constexpr float LN_EPS    = 1e-5f;
constexpr float LOG_EPS   = -13.815510557964274f;  // log(1e-6)

typedef __attribute__((ext_vector_type(16))) __bf16 v16bf;
typedef __attribute__((ext_vector_type(8)))  __bf16 v8bf;
typedef __attribute__((ext_vector_type(4)))  __bf16 v4bf;
typedef __attribute__((ext_vector_type(8)))  float  v8f;
typedef __attribute__((ext_vector_type(4)))  float  v4f;

__device__ __forceinline__ float gelu_exact(float x) {
    return 0.5f * x * (1.0f + erff(x * 0.70710678118654752f));
}

__device__ __forceinline__ float wave_sum(float v) {
    #pragma unroll
    for (int off = 16; off >= 1; off >>= 1) v += __shfl_xor(v, off, 32);
    return v;
}
__device__ __forceinline__ float wave_max(float v) {
    #pragma unroll
    for (int off = 16; off >= 1; off >>= 1) v = fmaxf(v, __shfl_xor(v, off, 32));
    return v;
}

// ---------------- generic bf16-WMMA GEMM ----------------
// out[M,N] = epilogue( A[M,K] @ B + bias ),  B is [K,N] (BT=false) or [N,K] (BT=true)
// epilogue: v = acc + bias; if GELU gelu(v); v *= rowscale[m]; v += res[m,n]
// K must be a multiple of 32 (all call sites: 768 or 3072).
// Block tile: 128(M) x 64(N) x 32(K); 8 waves, each owns a 32x32 output
// (4 f32 accumulators -> 4 v_wmma_f32_16x16x32_bf16 per K-step).
// Out-of-range rows/cols are CLAMPED at load (never branched): garbage only
// lands in accumulator slots whose stores are guarded, so no divergence and
// no zero-fill is needed in the hot loop.
template<bool GELU, bool BT>
__global__ __launch_bounds__(256)
void gemm_bf16_wmma(const float* __restrict__ A, const float* __restrict__ Bm,
                    const float* __restrict__ bias, const float* __restrict__ res,
                    const float* __restrict__ rowscale, float* __restrict__ out,
                    int M, int N, int K)
{
    __shared__ __align__(16) __bf16 As[128][32];  // 8 KB
    __shared__ __align__(16) __bf16 Bt[64][32];   // 4 KB (always N-major / transposed)

    const int tid  = threadIdx.x;
    const int lane = tid & 31;
    const int wave = tid >> 5;
    const int wm   = wave & 3;          // 4 waves along M (32 rows each)
    const int wn   = wave >> 2;         // 2 waves along N (32 cols each)
    const int half = lane >> 4;
    const int l16  = lane & 15;
    const int m0   = blockIdx.y * 128;
    const int n0   = blockIdx.x * 64;
    const bool nfull = (N - n0) >= 64;  // block-uniform: fast vectorized B staging

    v8f acc00 = {}, acc01 = {}, acc10 = {}, acc11 = {};

    for (int k0 = 0; k0 < K; k0 += 32) {
        __syncthreads();
        // ---- stage A: 128x32 fp32 -> bf16, float4 loads, row-clamped ----
        #pragma unroll
        for (int i = 0; i < 4; ++i) {            // 1024 vec4 = 4096 elems
            int v  = tid + i * 256;
            int r  = v >> 3;
            int c4 = (v & 7) * 4;
            int gr = m0 + r; gr = gr < M ? gr : M - 1;
            v4f av = *reinterpret_cast<const v4f*>(A + (size_t)gr * K + k0 + c4);
            v4bf ab = { (__bf16)av[0], (__bf16)av[1], (__bf16)av[2], (__bf16)av[3] };
            *reinterpret_cast<v4bf*>(&As[r][c4]) = ab;
        }
        // ---- stage B into Bt[n][k] ----
        if (BT) {
            // weight stored [N,K]: vectorize along K, clamp row
            #pragma unroll
            for (int i = 0; i < 2; ++i) {        // 512 vec4 = 2048 elems
                int v  = tid + i * 256;
                int n  = v >> 3;
                int c4 = (v & 7) * 4;
                int gn = n0 + n; gn = gn < N ? gn : N - 1;
                v4f bv = *reinterpret_cast<const v4f*>(Bm + (size_t)gn * K + k0 + c4);
                v4bf bb = { (__bf16)bv[0], (__bf16)bv[1], (__bf16)bv[2], (__bf16)bv[3] };
                *reinterpret_cast<v4bf*>(&Bt[n][c4]) = bb;
            }
        } else if (nfull) {
            // weight stored [K,N], full 64-col block: vectorize along N, scatter to LDS
            #pragma unroll
            for (int i = 0; i < 2; ++i) {
                int v  = tid + i * 256;
                int c  = v >> 4;
                int n4 = (v & 15) * 4;
                v4f bv = *reinterpret_cast<const v4f*>(Bm + (size_t)(k0 + c) * N + n0 + n4);
                Bt[n4 + 0][c] = (__bf16)bv[0];
                Bt[n4 + 1][c] = (__bf16)bv[1];
                Bt[n4 + 2][c] = (__bf16)bv[2];
                Bt[n4 + 3][c] = (__bf16)bv[3];
            }
        } else {
            // tail block in N (head GEMM only): scalar, col-clamped, still uniform
            #pragma unroll
            for (int i = 0; i < 8; ++i) {
                int l = tid + i * 256;
                int n = l >> 5;
                int c = l & 31;
                int gn = n0 + n; gn = gn < N ? gn : N - 1;
                Bt[n][c] = (__bf16)Bm[(size_t)(k0 + c) * N + gn];
            }
        }
        __syncthreads();

        // ---- fragments (layouts per CDNA5 ISA 7.12.2) ----
        // A 16x32 bf16: lane(h,l16) row = tile row l16; K = {h*8..+7, 16+h*8..+7}
        const int ar0 = wm * 32 + l16;
        const int ar1 = ar0 + 16;
        v8bf a0lo = *reinterpret_cast<const v8bf*>(&As[ar0][half * 8]);
        v8bf a0hi = *reinterpret_cast<const v8bf*>(&As[ar0][16 + half * 8]);
        v16bf af0 = __builtin_shufflevector(a0lo, a0hi,
            0,1,2,3,4,5,6,7,8,9,10,11,12,13,14,15);
        v8bf a1lo = *reinterpret_cast<const v8bf*>(&As[ar1][half * 8]);
        v8bf a1hi = *reinterpret_cast<const v8bf*>(&As[ar1][16 + half * 8]);
        v16bf af1 = __builtin_shufflevector(a1lo, a1hi,
            0,1,2,3,4,5,6,7,8,9,10,11,12,13,14,15);
        // B 32x16 bf16: lane(h,l16) col = l16; K = {h*8..+7, 16+h*8..+7}
        const int br0 = wn * 32 + l16;
        const int br1 = br0 + 16;
        v8bf b0lo = *reinterpret_cast<const v8bf*>(&Bt[br0][half * 8]);
        v8bf b0hi = *reinterpret_cast<const v8bf*>(&Bt[br0][16 + half * 8]);
        v16bf bf0 = __builtin_shufflevector(b0lo, b0hi,
            0,1,2,3,4,5,6,7,8,9,10,11,12,13,14,15);
        v8bf b1lo = *reinterpret_cast<const v8bf*>(&Bt[br1][half * 8]);
        v8bf b1hi = *reinterpret_cast<const v8bf*>(&Bt[br1][16 + half * 8]);
        v16bf bf1 = __builtin_shufflevector(b1lo, b1hi,
            0,1,2,3,4,5,6,7,8,9,10,11,12,13,14,15);

        acc00 = __builtin_amdgcn_wmma_f32_16x16x32_bf16(false, af0, false, bf0, (short)0, acc00, false, false);
        acc01 = __builtin_amdgcn_wmma_f32_16x16x32_bf16(false, af0, false, bf1, (short)0, acc01, false, false);
        acc10 = __builtin_amdgcn_wmma_f32_16x16x32_bf16(false, af1, false, bf0, (short)0, acc10, false, false);
        acc11 = __builtin_amdgcn_wmma_f32_16x16x32_bf16(false, af1, false, bf1, (short)0, acc11, false, false);
    }

    // ---- epilogue: D element r -> row (r + 8*half), col l16 within 16x16 tile ----
    #pragma unroll
    for (int mt = 0; mt < 2; ++mt) {
        #pragma unroll
        for (int nt = 0; nt < 2; ++nt) {
            const v8f& acc = mt == 0 ? (nt == 0 ? acc00 : acc01)
                                     : (nt == 0 ? acc10 : acc11);
            int gn = n0 + wn * 32 + nt * 16 + l16;
            if (gn >= N) continue;
            float bs = bias ? bias[gn] : 0.0f;
            #pragma unroll
            for (int r = 0; r < 8; ++r) {
                int gm = m0 + wm * 32 + mt * 16 + r + 8 * half;
                if (gm >= M) continue;
                float v = acc[r] + bs;
                if (GELU) v = gelu_exact(v);
                if (rowscale) v *= rowscale[gm];
                if (res) v += res[(size_t)gm * N + gn];
                out[(size_t)gm * N + gn] = v;
            }
        }
    }
}

// ---------------- patch embedding helpers ----------------
// Build A_patch[B*196, 768]: A[row, c*256 + p*16 + q] = images[b, c, gx*16+p, gy*16+q]
__global__ __launch_bounds__(256)
void patchify_kernel(const float* __restrict__ images, float* __restrict__ Ap)
{
    int idx = blockIdx.x * 256 + threadIdx.x;
    int total = BB * NPAT * DM;
    if (idx >= total) return;
    int row = idx / DM, k = idx % DM;
    int b = row / NPAT, t = row % NPAT;
    int gx = t / GG, gy = t % GG;
    int c = k >> 8, rr = k & 255;
    int p = rr >> 4, q = rr & 15;
    Ap[idx] = images[((size_t)(b * CC + c) * IMG + gx * PP + p) * IMG + gy * PP + q];
}

// x[b,0,:] = cls + pos[0];  x[b,1+t,:] = xt[b*196+t] + pos[1+t]
__global__ __launch_bounds__(256)
void assemble_kernel(const float* __restrict__ xt, const float* __restrict__ cls,
                     const float* __restrict__ pos, float* __restrict__ x)
{
    int idx = blockIdx.x * 256 + threadIdx.x;
    int total = TOKS * DM;
    if (idx >= total) return;
    int row = idx / DM, d = idx % DM;
    int b = row / NTOK, n = row % NTOK;
    float v = (n == 0) ? cls[d] : xt[(size_t)(b * NPAT + n - 1) * DM + d];
    x[idx] = v + pos[(size_t)n * DM + d];
}

// ---------------- layernorm: one wave per row (D=768 -> 24/lane) ----------------
__global__ __launch_bounds__(256)
void layernorm_kernel(const float* __restrict__ in, long inStride,
                      float* __restrict__ out, long outStride,
                      const float* __restrict__ g, const float* __restrict__ bta, int M)
{
    const int lane = threadIdx.x & 31;
    const int wave = threadIdx.x >> 5;
    const int row  = blockIdx.x * 8 + wave;
    if (row >= M) return;
    const float* xr = in + (size_t)row * inStride;
    float v[24];
    float s = 0.0f;
    #pragma unroll
    for (int t = 0; t < 24; ++t) { v[t] = xr[lane + t * 32]; s += v[t]; }
    s = wave_sum(s);
    float mu = s * (1.0f / DM);
    float var = 0.0f;
    #pragma unroll
    for (int t = 0; t < 24; ++t) { float d = v[t] - mu; var += d * d; }
    var = wave_sum(var) * (1.0f / DM);
    float rstd = rsqrtf(var + LN_EPS);
    float* orow = out + (size_t)row * outStride;
    #pragma unroll
    for (int t = 0; t < 24; ++t) {
        int c = lane + t * 32;
        orow[c] = (v[t] - mu) * rstd * g[c] + bta[c];
    }
}

// ---------------- attention: scores + softmax, one wave per (b,h,q) ----------------
// qkv layout: [B*N, 3*768], q at +0, k at +768, v at +1536 (within: h*64+d)
__global__ __launch_bounds__(256)
void attn_scores_kernel(const float* __restrict__ qkv, const float* __restrict__ addmask,
                        float* __restrict__ probs)
{
    __shared__ float qs[8][HD];
    const int lane = threadIdx.x & 31;
    const int wave = threadIdx.x >> 5;
    const int wid  = blockIdx.x * 8 + wave;         // grid chosen exact: B*NH*NTOK / 8
    const int q = wid % NTOK;
    const int h = (wid / NTOK) % NH;
    const int b = wid / (NTOK * NH);

    const float* qrow = qkv + (size_t)(b * NTOK + q) * (3 * DM) + h * HD;
    qs[wave][lane]      = qrow[lane];
    qs[wave][lane + 32] = qrow[lane + 32];
    __syncthreads();

    float s[7];
    #pragma unroll
    for (int t = 0; t < 7; ++t) {
        int j = lane + t * 32;
        float val = -__builtin_inff();
        if (j < NTOK) {
            const float* krow = qkv + (size_t)(b * NTOK + j) * (3 * DM) + DM + h * HD;
            float dot = 0.0f;
            #pragma unroll
            for (int d = 0; d < HD; ++d) dot += qs[wave][d] * krow[d];
            val = dot * ATT_SCALE;
            if (addmask) val += addmask[b * NTOK + j];
        }
        s[t] = val;
    }
    float mx = s[0];
    #pragma unroll
    for (int t = 1; t < 7; ++t) mx = fmaxf(mx, s[t]);
    mx = wave_max(mx);
    float p[7];
    float sum = 0.0f;
    #pragma unroll
    for (int t = 0; t < 7; ++t) {
        int j = lane + t * 32;
        p[t] = (j < NTOK) ? expf(s[t] - mx) : 0.0f;
        sum += p[t];
    }
    sum = wave_sum(sum);
    float inv = 1.0f / sum;
    float* prow = probs + (size_t)((b * NH + h) * NTOK + q) * NTOK;
    #pragma unroll
    for (int t = 0; t < 7; ++t) {
        int j = lane + t * 32;
        if (j < NTOK) prow[j] = p[t] * inv;
    }
}

// P @ V : one wave per (b,q,h); lane owns d = lane, lane+32
__global__ __launch_bounds__(256)
void attn_pv_kernel(const float* __restrict__ probs, const float* __restrict__ qkv,
                    float* __restrict__ out)
{
    const int lane = threadIdx.x & 31;
    const int wave = threadIdx.x >> 5;
    const int wid  = blockIdx.x * 8 + wave;
    const int h = wid % NH;
    const int q = (wid / NH) % NTOK;
    const int b = wid / (NH * NTOK);

    const float* prow = probs + (size_t)((b * NH + h) * NTOK + q) * NTOK;
    float a0 = 0.0f, a1 = 0.0f;
    for (int j = 0; j < NTOK; ++j) {
        float pv = prow[j];
        const float* vrow = qkv + (size_t)(b * NTOK + j) * (3 * DM) + 2 * DM + h * HD;
        a0 += pv * vrow[lane];
        a1 += pv * vrow[lane + 32];
    }
    float* orow = out + (size_t)(b * NTOK + q) * DM + h * HD;
    orow[lane]      = a0;
    orow[lane + 32] = a1;
}

// ---------------- gate logits: one wave per token (dot over 128) ----------------
__global__ __launch_bounds__(256)
void gate_logits_kernel(const float* __restrict__ gh, const float* __restrict__ w2,
                        const float* __restrict__ b2, float* __restrict__ logits, int M)
{
    const int lane = threadIdx.x & 31;
    const int wave = threadIdx.x >> 5;
    const int row  = blockIdx.x * 8 + wave;
    if (row >= M) return;
    float dot = 0.0f;
    #pragma unroll
    for (int t = 0; t < 4; ++t) {
        int c = lane + t * 32;
        dot += gh[(size_t)row * GHD + c] * w2[c];
    }
    dot = wave_sum(dot);
    if (lane == 0) logits[row] = dot + b2[0];
}

// ---------------- top-k mask: one block per batch, exact rank counting ----------------
__global__ __launch_bounds__(256)
void topk_mask_kernel(const float* __restrict__ logits, float* __restrict__ m,
                      float* __restrict__ am)
{
    __shared__ float lg[NPAT];
    const int b = blockIdx.x;
    const int tid = threadIdx.x;
    if (tid < NPAT) lg[tid] = logits[b * NTOK + 1 + tid];
    __syncthreads();
    if (tid < NPAT) {
        float li = lg[tid];
        int rank = 0;
        for (int j = 0; j < NPAT; ++j) {
            float lj = lg[j];
            rank += (lj > li) || (lj == li && j < tid);   // top_k tie-break: lower index first
        }
        bool keep = rank < TOPK;
        m[b * NTOK + 1 + tid]  = keep ? 1.0f : 0.0f;
        am[b * NTOK + 1 + tid] = keep ? 0.0f : LOG_EPS;
    }
    if (tid == 0) { m[b * NTOK] = 1.0f; am[b * NTOK] = 0.0f; }  // cls always kept
}

// ---------------- host-side orchestration ----------------
template<bool GELU, bool BT>
static void launch_gemm(const float* A, const float* Bm, const float* bias,
                        const float* res, const float* scale, float* out,
                        int M, int N, int K, hipStream_t s)
{
    dim3 grid((N + 63) / 64, (M + 127) / 128);
    gemm_bf16_wmma<GELU, BT><<<grid, dim3(256), 0, s>>>(A, Bm, bias, res, scale, out, M, N, K);
}

extern "C" void kernel_launch(void* const* d_in, const int* in_sizes, int n_in,
                              void* d_out, int out_size, void* d_ws, size_t ws_size,
                              hipStream_t stream)
{
    (void)in_sizes; (void)n_in; (void)out_size; (void)ws_size;
    int a = 0;
    const float* images   = (const float*)d_in[a++];
    const float* patch_w  = (const float*)d_in[a++];
    const float* patch_b  = (const float*)d_in[a++];
    const float* cls_tok  = (const float*)d_in[a++];
    const float* pos_emb  = (const float*)d_in[a++];
    const float* w_ln1_g  = (const float*)d_in[a++];
    const float* w_ln1_b  = (const float*)d_in[a++];
    const float* w_qkv_w  = (const float*)d_in[a++];
    const float* w_qkv_b  = (const float*)d_in[a++];
    const float* w_proj_w = (const float*)d_in[a++];
    const float* w_proj_b = (const float*)d_in[a++];
    const float* w_ln2_g  = (const float*)d_in[a++];
    const float* w_ln2_b  = (const float*)d_in[a++];
    const float* w_fc1_w  = (const float*)d_in[a++];
    const float* w_fc1_b  = (const float*)d_in[a++];
    const float* w_fc2_w  = (const float*)d_in[a++];
    const float* w_fc2_b  = (const float*)d_in[a++];
    const float* d_ln1_g  = (const float*)d_in[a++];
    const float* d_ln1_b  = (const float*)d_in[a++];
    const float* d_qkv_w  = (const float*)d_in[a++];
    const float* d_qkv_b  = (const float*)d_in[a++];
    const float* d_proj_w = (const float*)d_in[a++];
    const float* d_proj_b = (const float*)d_in[a++];
    const float* d_ln2_g  = (const float*)d_in[a++];
    const float* d_ln2_b  = (const float*)d_in[a++];
    const float* d_fc1_w  = (const float*)d_in[a++];
    const float* d_fc1_b  = (const float*)d_in[a++];
    const float* d_fc2_w  = (const float*)d_in[a++];
    const float* d_fc2_b  = (const float*)d_in[a++];
    const float* d_gln_g  = (const float*)d_in[a++];
    const float* d_gln_b  = (const float*)d_in[a++];
    const float* d_gw1    = (const float*)d_in[a++];
    const float* d_gb1    = (const float*)d_in[a++];
    const float* d_gw2    = (const float*)d_in[a++];
    const float* d_gb2    = (const float*)d_in[a++];
    const float* norm_g   = (const float*)d_in[a++];
    const float* norm_b   = (const float*)d_in[a++];
    const float* head_w   = (const float*)d_in[a++];
    const float* head_b   = (const float*)d_in[a++];

    float* ws = (float*)d_ws;
    size_t o = 0;
    float* x       = ws + o; o += (size_t)TOKS * DM;        // 4.84M
    float* hbuf    = ws + o; o += (size_t)TOKS * DM;        // 4.84M
    float* qkv     = ws + o; o += (size_t)TOKS * 3 * DM;    // 14.5M
    float* attnout = ws + o; o += (size_t)TOKS * DM;        // 4.84M
    float* big     = ws + o; o += (size_t)TOKS * HID;       // 19.4M (probs/mlph/patchA alias)
    float* gateh   = ws + o; o += (size_t)TOKS * GHD;
    float* logits  = ws + o; o += (size_t)TOKS;
    float* maskm   = ws + o; o += (size_t)BB * NTOK;
    float* addmask = ws + o; o += (size_t)BB * NTOK;
    float* clsbuf  = ws + o; o += (size_t)BB * DM;
    float* probs  = big;                 // B*NH*197*197 = 14.9M  (attention phase)
    float* mlph   = big;                 // TOKS*HID      = 19.4M  (MLP phase)
    float* patchA = big;                 // B*196*768     = 4.8M   (patch phase)
    float* xt     = attnout;             // patch-GEMM output (6272 rows <= 6304)

    const int AW_GRID = (BB * NH * NTOK) / 8;   // 9456, exact
    const int LN_GRID = TOKS / 8;               // 788, exact

    // ---- patch embedding ----
    {
        int np = BB * NPAT * DM;
        patchify_kernel<<<(np + 255) / 256, 256, 0, stream>>>(images, patchA);
        // patch_w is (D, C*P*P) = [N,K] row-major -> BT=true
        launch_gemm<false, true>(patchA, patch_w, patch_b, nullptr, nullptr, xt,
                                 BB * NPAT, DM, DM, stream);
        int nx = TOKS * DM;
        assemble_kernel<<<(nx + 255) / 256, 256, 0, stream>>>(xt, cls_tok, pos_emb, x);
    }

    // ---- warm layers (no mask) ----
    for (int i = 0; i < NWARM; ++i) {
        layernorm_kernel<<<LN_GRID, 256, 0, stream>>>(x, DM, hbuf, DM,
            w_ln1_g + (size_t)i * DM, w_ln1_b + (size_t)i * DM, TOKS);
        launch_gemm<false, false>(hbuf, w_qkv_w + (size_t)i * DM * 3 * DM,
            w_qkv_b + (size_t)i * 3 * DM, nullptr, nullptr, qkv, TOKS, 3 * DM, DM, stream);
        attn_scores_kernel<<<AW_GRID, 256, 0, stream>>>(qkv, nullptr, probs);
        attn_pv_kernel<<<AW_GRID, 256, 0, stream>>>(probs, qkv, attnout);
        launch_gemm<false, false>(attnout, w_proj_w + (size_t)i * DM * DM,
            w_proj_b + (size_t)i * DM, x, nullptr, x, TOKS, DM, DM, stream);
        layernorm_kernel<<<LN_GRID, 256, 0, stream>>>(x, DM, hbuf, DM,
            w_ln2_g + (size_t)i * DM, w_ln2_b + (size_t)i * DM, TOKS);
        launch_gemm<true, false>(hbuf, w_fc1_w + (size_t)i * DM * HID,
            w_fc1_b + (size_t)i * HID, nullptr, nullptr, mlph, TOKS, HID, DM, stream);
        launch_gemm<false, false>(mlph, w_fc2_w + (size_t)i * HID * DM,
            w_fc2_b + (size_t)i * DM, x, nullptr, x, TOKS, DM, HID, stream);
    }

    // ---- dynamic layers (gated + masked) ----
    for (int j = 0; j < NDYN; ++j) {
        // gate network -> mask
        layernorm_kernel<<<LN_GRID, 256, 0, stream>>>(x, DM, hbuf, DM,
            d_gln_g + (size_t)j * DM, d_gln_b + (size_t)j * DM, TOKS);
        launch_gemm<true, false>(hbuf, d_gw1 + (size_t)j * DM * GHD,
            d_gb1 + (size_t)j * GHD, nullptr, nullptr, gateh, TOKS, GHD, DM, stream);
        gate_logits_kernel<<<LN_GRID, 256, 0, stream>>>(gateh,
            d_gw2 + (size_t)j * GHD, d_gb2 + j, logits, TOKS);
        topk_mask_kernel<<<BB, 256, 0, stream>>>(logits, maskm, addmask);

        // masked attention
        layernorm_kernel<<<LN_GRID, 256, 0, stream>>>(x, DM, hbuf, DM,
            d_ln1_g + (size_t)j * DM, d_ln1_b + (size_t)j * DM, TOKS);
        launch_gemm<false, false>(hbuf, d_qkv_w + (size_t)j * DM * 3 * DM,
            d_qkv_b + (size_t)j * 3 * DM, nullptr, nullptr, qkv, TOKS, 3 * DM, DM, stream);
        attn_scores_kernel<<<AW_GRID, 256, 0, stream>>>(qkv, addmask, probs);
        attn_pv_kernel<<<AW_GRID, 256, 0, stream>>>(probs, qkv, attnout);
        launch_gemm<false, false>(attnout, d_proj_w + (size_t)j * DM * DM,
            d_proj_b + (size_t)j * DM, x, nullptr, x, TOKS, DM, DM, stream);

        // masked MLP: x = x + mlp(ln(x)) * m[:, :, None]
        layernorm_kernel<<<LN_GRID, 256, 0, stream>>>(x, DM, hbuf, DM,
            d_ln2_g + (size_t)j * DM, d_ln2_b + (size_t)j * DM, TOKS);
        launch_gemm<true, false>(hbuf, d_fc1_w + (size_t)j * DM * HID,
            d_fc1_b + (size_t)j * HID, nullptr, nullptr, mlph, TOKS, HID, DM, stream);
        launch_gemm<false, false>(mlph, d_fc2_w + (size_t)j * HID * DM,
            d_fc2_b + (size_t)j * DM, x, maskm, x, TOKS, DM, HID, stream);
    }

    // ---- final LN on cls rows + classifier head ----
    layernorm_kernel<<<(BB + 7) / 8, 256, 0, stream>>>(x, (long)NTOK * DM,
        clsbuf, DM, norm_g, norm_b, BB);
    launch_gemm<false, false>(clsbuf, head_w, head_b, nullptr, nullptr,
                              (float*)d_out, BB, NCLS, DM, stream);
}
